// HSTUBlock_24232205484713
// MI455X (gfx1250) — matrix operand order repaired and checked by reading, not compiled, not run on hardware
//
#include <hip/hip_runtime.h>
#include <hip/hip_bf16.h>

// ---------------- problem constants ----------------
#define BB 4
#define NN 2048
#define DD 512
#define HH 8
#define DVV 64
#define EE 2048           // uvqk output columns
#define NBUCK 128
#define LNEPS 1e-5f

#define ASYNC_LDS 1       // use global_load_async_to_lds_b128 for tile staging

typedef _Float16 v8h  __attribute__((ext_vector_type(8)));
typedef _Float16 v16h __attribute__((ext_vector_type(16)));
typedef float    v8f  __attribute__((ext_vector_type(8)));

// ---------------- WMMA helpers ----------------
__device__ __forceinline__ v8f wmma_f16(v16h a, v16h b, v8f c) {
  return __builtin_amdgcn_wmma_f32_16x16x32_f16(
      false, a, false, b, (short)0, c, false, false);
}

__device__ __forceinline__ v16h frag_from(const _Float16* lo, const _Float16* hi) {
  union { v16h v; v8h h[2]; } u;
  u.h[0] = *(const v8h*)lo;
  u.h[1] = *(const v8h*)hi;
  return u.v;
}

// A fragment: row-major [16 rows][32 K], `stride` halves.
// lane L (m=L&15,g=L>>4): halves0..7 = K 8g..8g+7, halves8..15 = K 16+8g..23+8g
__device__ __forceinline__ v16h a_frag(const _Float16* base, int stride, int lane) {
  int m = lane & 15, g = lane >> 4;
  const _Float16* p = base + m * stride;
  return frag_from(p + 8 * g, p + 16 + 8 * g);
}

// B fragment: [16 N-cols][K] layout, `stride` halves, K window at koff.
// lane L (n=L&15,g=L>>4): halves0..7 = K 8g..8g+7, halves8..15 = K 16+8g..23+8g
__device__ __forceinline__ v16h b_frag(const _Float16* base, int stride, int koff, int lane) {
  int n = lane & 15, g = lane >> 4;
  const _Float16* p = base + n * stride + koff;
  return frag_from(p + 8 * g, p + 16 + 8 * g);
}

__device__ __forceinline__ float silu_f(float x) {
  return x / (1.0f + __expf(-x));
}

// ---------------- async global->LDS staging (CDNA5 ASYNCcnt path) ------------
__device__ __forceinline__ unsigned ldsOff(const void* p) {
  return (unsigned)(unsigned long long)
      (__attribute__((address_space(3))) const void*)p;
}

__device__ __forceinline__ void asyncLoadB128(void* ldsDst, const void* gSrc) {
#if ASYNC_LDS
  asm volatile("global_load_async_to_lds_b128 %0, %1, off"
               :: "v"(ldsOff(ldsDst)), "v"(gSrc)
               : "memory");
#else
  *(v8h*)ldsDst = *(const v8h*)gSrc;
#endif
}

// wait until per-wave ASYNCcnt <= N (async loads complete in issue order)
template <int N>
__device__ __forceinline__ void asyncWaitLE() {
#if ASYNC_LDS
  asm volatile("s_wait_asynccnt %0" :: "i"(N) : "memory");
#endif
}

// ---------------- kernel: f32 [R][C] -> f16 [C][R] transpose (weights) --------
__global__ __launch_bounds__(256) void transpose_f32f16_kernel(
    const float* __restrict__ src, _Float16* __restrict__ dst, int R, int C) {
  __shared__ float tile[64][65];
  int c0 = blockIdx.x * 64, r0 = blockIdx.y * 64;
  int t = threadIdx.x;
#pragma unroll
  for (int i = 0; i < 16; ++i) {
    int e = t + i * 256;
    int r = e >> 6, c = e & 63;
    tile[r][c] = src[(size_t)(r0 + r) * C + c0 + c];
  }
  __syncthreads();
#pragma unroll
  for (int i = 0; i < 16; ++i) {
    int e = t + i * 256;
    int n = e >> 6, k = e & 63;
    dst[(size_t)(c0 + n) * R + r0 + k] = (_Float16)tile[k][n];
  }
}

// ---------------- row LayerNorm (D=512), optional elementwise gate -> f16 ----
__global__ __launch_bounds__(256) void ln_rows_kernel(
    const float* __restrict__ x, const float* __restrict__ w,
    const float* __restrict__ bias, const float* __restrict__ umul,
    _Float16* __restrict__ out) {
  __shared__ float part[16];
  __shared__ float red[2];
  int row = blockIdx.x;
  const float* xr = x + (size_t)row * DD;
  int t = threadIdx.x;
  float v0 = xr[t], v1 = xr[t + 256];
  float s = v0 + v1, s2 = v0 * v0 + v1 * v1;
#pragma unroll
  for (int off = 16; off > 0; off >>= 1) {
    s  += __shfl_down(s,  off, 32);
    s2 += __shfl_down(s2, off, 32);
  }
  if ((t & 31) == 0) { part[t >> 5] = s; part[8 + (t >> 5)] = s2; }
  __syncthreads();
  if (t == 0) {
    float ts = 0.f, t2 = 0.f;
#pragma unroll
    for (int i = 0; i < 8; ++i) { ts += part[i]; t2 += part[8 + i]; }
    float mu = ts * (1.0f / DD);
    red[0] = mu;
    red[1] = rsqrtf(t2 * (1.0f / DD) - mu * mu + LNEPS);
  }
  __syncthreads();
  float mu = red[0], rs = red[1];
  float y0 = (v0 - mu) * rs * w[t] + bias[t];
  float y1 = (v1 - mu) * rs * w[t + 256] + bias[t + 256];
  if (umul) {
    const float* ur = umul + (size_t)row * DD;
    y0 *= ur[t];
    y1 *= ur[t + 256];
  }
  out[(size_t)row * DD + t]       = (_Float16)y0;
  out[(size_t)row * DD + t + 256] = (_Float16)y1;
}

// -------- stage one 64x32 A-tile + 128x32 B-tile (3 async ops / thread) ------
__device__ __forceinline__ void stage_gemm_tile(
    _Float16* lA, _Float16* lB, const _Float16* amat, const _Float16* bmat,
    int m0, int n0, int kk, int t) {
  {
    int e = t * 8, r = e >> 5, c = e & 31;
    asyncLoadB128(&lA[r * 40 + c], amat + (size_t)(m0 + r) * DD + kk + c);
  }
  {
    int e = t * 16, n = e >> 5, c = e & 31;
    const _Float16* g0 = bmat + (size_t)(n0 + n) * DD + kk + c;
    asyncLoadB128(&lB[n * 40 + c],     g0);
    asyncLoadB128(&lB[n * 40 + c + 8], g0 + 8);
  }
}

// ---------------- uvqk GEMM: xn(f16) @ wt(f16,[E][D]) , SiLU, split+route ----
// grid (M/64, E/128), block 256 = 8 waves (2 M x 4 N). Wave tile 32x32.
// Double-buffered async LDS pipeline.
__global__ __launch_bounds__(256) void uvqk_gemm_kernel(
    const _Float16* __restrict__ xn, const _Float16* __restrict__ wt,
    float* __restrict__ u_out, _Float16* __restrict__ qf,
    _Float16* __restrict__ kf, _Float16* __restrict__ vt) {
  __shared__ __align__(32) _Float16 lA[2][64 * 40];
  __shared__ __align__(32) _Float16 lB[2][128 * 40];
  int m0 = blockIdx.x * 64;
  int n0 = blockIdx.y * 128;
  int t = threadIdx.x, lane = t & 31, wv = t >> 5;
  int wm = wv & 1, wn = wv >> 1;
  v8f acc[2][2] = {};
  stage_gemm_tile(lA[0], lB[0], xn, wt, m0, n0, 0, t);
  int buf = 0;
  for (int kk = 0; kk < DD; kk += 32) {
    bool more = (kk + 32) < DD;
    if (more) {
      stage_gemm_tile(lA[buf ^ 1], lB[buf ^ 1], xn, wt, m0, n0, kk + 32, t);
      asyncWaitLE<3>();          // current buffer's 3 ops complete
    } else {
      asyncWaitLE<0>();
    }
    __syncthreads();
    const _Float16* pA = lA[buf];
    const _Float16* pB = lB[buf];
    v16h a0 = a_frag(&pA[(wm * 32) * 40],      40, lane);
    v16h a1 = a_frag(&pA[(wm * 32 + 16) * 40], 40, lane);
    v16h b0 = b_frag(&pB[(wn * 32) * 40],      40, 0, lane);
    v16h b1 = b_frag(&pB[(wn * 32 + 16) * 40], 40, 0, lane);
    acc[0][0] = wmma_f16(a0, b0, acc[0][0]);
    acc[0][1] = wmma_f16(a0, b1, acc[0][1]);
    acc[1][0] = wmma_f16(a1, b0, acc[1][0]);
    acc[1][1] = wmma_f16(a1, b1, acc[1][1]);
    buf ^= 1;
    __syncthreads();             // protects buffer reuse two iterations out
  }
  int g = lane >> 4, nl = lane & 15;
#pragma unroll
  for (int mi = 0; mi < 2; ++mi)
#pragma unroll
    for (int ni = 0; ni < 2; ++ni) {
      int gc = n0 + wn * 32 + ni * 16 + nl;
#pragma unroll
      for (int r = 0; r < 8; ++r) {
        int gr = m0 + wm * 32 + mi * 16 + 8 * g + r;
        float v = silu_f(acc[mi][ni][r]);
        int b = gr >> 11, nrow = gr & (NN - 1);
        if (gc < 512) {
          u_out[(size_t)gr * 512 + gc] = v;
        } else if (gc < 1024) {               // V -> [b][h][d][n]
          int c = gc - 512, h = c >> 6, d = c & 63;
          vt[(((size_t)b * HH + h) * DVV + d) * NN + nrow] = (_Float16)v;
        } else if (gc < 1536) {               // Q [b][h][n][d]
          int c = gc - 1024, h = c >> 6, d = c & 63;
          qf[(((size_t)b * HH + h) * NN + nrow) * 64 + d] = (_Float16)v;
        } else {                              // K [b][h][n][d]
          int c = gc - 1536, h = c >> 6, d = c & 63;
          kf[(((size_t)b * HH + h) * NN + nrow) * 64 + d] = (_Float16)v;
        }
      }
    }
}

// -------- stage one K(32x64) + V^T(64x32) tile (4 async ops / thread) --------
__device__ __forceinline__ void stage_attn_tile(
    _Float16* lK, _Float16* lV, const _Float16* kptr, const _Float16* vptr,
    int m0, int t) {
  {
    int mm = t >> 2, ds_ = (t & 3) * 16;
    const _Float16* g0 = kptr + (size_t)(m0 + mm) * 64 + ds_;
    asyncLoadB128(&lK[mm * 72 + ds_],     g0);
    asyncLoadB128(&lK[mm * 72 + ds_ + 8], g0 + 8);
  }
  {
    int d = t >> 1, ms = (t & 1) * 16;
    const _Float16* g0 = vptr + (size_t)d * NN + m0 + ms;
    asyncLoadB128(&lV[d * 40 + ms],     g0);
    asyncLoadB128(&lV[d * 40 + ms + 8], g0 + 8);
  }
}

// ---------------- fused causal HSTU attention ----------------
// grid (N/64, H, B), block 128 = 4 waves; wave handles 16 query rows.
// Double-buffered async LDS pipeline over 32-key tiles.
__global__ __launch_bounds__(128) void attn_kernel(
    const _Float16* __restrict__ qf, const _Float16* __restrict__ kf,
    const _Float16* __restrict__ vt, const int* __restrict__ tsp,
    const unsigned char* __restrict__ pad, const float* __restrict__ ts_w,
    const float* __restrict__ pos_w, float* __restrict__ attn) {
  __shared__ __align__(32) _Float16 lK[2][32 * 72];
  __shared__ __align__(32) _Float16 lV[2][64 * 40];
  __shared__ __align__(32) _Float16 lS[4][16 * 40];
  int blk = blockIdx.x, h = blockIdx.y, b = blockIdx.z;
  int t = threadIdx.x, lane = t & 31, wv = t >> 5;
  int r0 = blk * 64 + wv * 16;
  size_t bh = (size_t)b * HH + h;
  const _Float16* qptr = qf + bh * NN * 64;
  const _Float16* kptr = kf + bh * NN * 64;
  const _Float16* vptr = vt + bh * DVV * NN;
  int ml = lane & 15, g = lane >> 4;

  const _Float16* qrow = qptr + (size_t)(r0 + ml) * 64;
  v16h aq0 = frag_from(qrow + 8 * g,      qrow + 16 + 8 * g);
  v16h aq1 = frag_from(qrow + 32 + 8 * g, qrow + 48 + 8 * g);

  int tsrow[8];
#pragma unroll
  for (int r = 0; r < 8; ++r) {
    int i = r0 + 8 * g + r;
    int ii = (i + 1 < NN) ? i + 1 : NN - 1;
    tsrow[r] = tsp[b * NN + ii];
  }

  v8f o0 = {}, o1 = {}, o2 = {}, o3 = {};
  int mmax = blk * 64 + 64;
  stage_attn_tile(lK[0], lV[0], kptr, vptr, 0, t);
  int buf = 0;
  for (int m0 = 0; m0 < mmax; m0 += 32) {
    bool more = (m0 + 32) < mmax;
    if (more) {
      stage_attn_tile(lK[buf ^ 1], lV[buf ^ 1], kptr, vptr, m0 + 32, t);
      asyncWaitLE<4>();
    } else {
      asyncWaitLE<0>();
    }
    __syncthreads();
    if (m0 <= r0 + 15) {
      const _Float16* pK = lK[buf];
      const _Float16* pV = lV[buf];
      v8f s0 = {}, s1 = {};
      s0 = wmma_f16(aq0, b_frag(&pK[0], 72, 0, lane), s0);
      s0 = wmma_f16(aq1, b_frag(&pK[0], 72, 32, lane), s0);
      s1 = wmma_f16(aq0, b_frag(&pK[16 * 72], 72, 0, lane), s1);
      s1 = wmma_f16(aq1, b_frag(&pK[16 * 72], 72, 32, lane), s1);
      _Float16* sptr = &lS[wv][0];
#pragma unroll
      for (int half = 0; half < 2; ++half) {
        v8f s = half ? s1 : s0;
        int j = m0 + half * 16 + ml;
        int tsj = tsp[b * NN + j];
        float padj = pad[b * NN + j] ? 0.0f : 1.0f;
#pragma unroll
        for (int r = 0; r < 8; ++r) {
          int i = r0 + 8 * g + r;
          float val = s[r] + pos_w[(NN - 1) + j - i];
          float ad = fmaxf(fabsf((float)(tsrow[r] - tsj)), 1.0f);
          int bucket = (int)(__logf(ad) * 3.3219281f);
          bucket = bucket < 0 ? 0 : (bucket > NBUCK ? NBUCK : bucket);
          val += ts_w[bucket];
          val = silu_f(val) * (1.0f / NN);
          if (j > i) val = 0.0f;
          val *= padj;
          sptr[(8 * g + r) * 40 + half * 16 + ml] = (_Float16)val;
        }
      }
      v16h as = frag_from(sptr + ml * 40 + 8 * g, sptr + ml * 40 + 16 + 8 * g);
      o0 = wmma_f16(as, b_frag(&pV[0],       40, 0, lane), o0);
      o1 = wmma_f16(as, b_frag(&pV[16 * 40], 40, 0, lane), o1);
      o2 = wmma_f16(as, b_frag(&pV[32 * 40], 40, 0, lane), o2);
      o3 = wmma_f16(as, b_frag(&pV[48 * 40], 40, 0, lane), o3);
    }
    buf ^= 1;
    __syncthreads();
  }
  float* abase = attn + ((size_t)b * NN) * DD + h * 64;
#pragma unroll
  for (int r = 0; r < 8; ++r) {
    int i = r0 + 8 * g + r;
    float* arow = abase + (size_t)i * DD + ml;
    arow[0]  = o0[r];
    arow[16] = o1[r];
    arow[32] = o2[r];
    arow[48] = o3[r];
  }
}

// ---------------- out GEMM: oin(f16) @ wot(f16,[D][D]) + o_b + x, pad --------
// grid (M/64, D/128), block 256 = 8 waves (2 M x 4 N). Wave tile 32x32.
__global__ __launch_bounds__(256) void out_gemm_kernel(
    const _Float16* __restrict__ oin, const _Float16* __restrict__ wot,
    const float* __restrict__ ob, const float* __restrict__ x,
    const unsigned char* __restrict__ pad, float* __restrict__ out) {
  __shared__ __align__(32) _Float16 lA[2][64 * 40];
  __shared__ __align__(32) _Float16 lB[2][128 * 40];
  int m0 = blockIdx.x * 64;
  int n0 = blockIdx.y * 128;
  int t = threadIdx.x, lane = t & 31, wv = t >> 5;
  int wm = wv & 1, wn = wv >> 1;
  v8f acc[2][2] = {};
  stage_gemm_tile(lA[0], lB[0], oin, wot, m0, n0, 0, t);
  int buf = 0;
  for (int kk = 0; kk < DD; kk += 32) {
    bool more = (kk + 32) < DD;
    if (more) {
      stage_gemm_tile(lA[buf ^ 1], lB[buf ^ 1], oin, wot, m0, n0, kk + 32, t);
      asyncWaitLE<3>();
    } else {
      asyncWaitLE<0>();
    }
    __syncthreads();
    const _Float16* pA = lA[buf];
    const _Float16* pB = lB[buf];
    v16h a0 = a_frag(&pA[(wm * 32) * 40],      40, lane);
    v16h a1 = a_frag(&pA[(wm * 32 + 16) * 40], 40, lane);
    v16h b0 = b_frag(&pB[(wn * 32) * 40],      40, 0, lane);
    v16h b1 = b_frag(&pB[(wn * 32 + 16) * 40], 40, 0, lane);
    acc[0][0] = wmma_f16(a0, b0, acc[0][0]);
    acc[0][1] = wmma_f16(a0, b1, acc[0][1]);
    acc[1][0] = wmma_f16(a1, b0, acc[1][0]);
    acc[1][1] = wmma_f16(a1, b1, acc[1][1]);
    buf ^= 1;
    __syncthreads();
  }
  int g = lane >> 4, nl = lane & 15;
#pragma unroll
  for (int mi = 0; mi < 2; ++mi)
#pragma unroll
    for (int ni = 0; ni < 2; ++ni) {
      int gc = n0 + wn * 32 + ni * 16 + nl;
#pragma unroll
      for (int r = 0; r < 8; ++r) {
        int gr = m0 + wm * 32 + mi * 16 + 8 * g + r;
        float v = acc[mi][ni][r] + ob[gc] + x[(size_t)gr * DD + gc];
        if (pad[gr]) v = 0.0f;
        out[(size_t)gr * DD + gc] = v;
      }
    }
}

// ---------------- launcher ----------------
extern "C" void kernel_launch(void* const* d_in, const int* in_sizes, int n_in,
                              void* d_out, int out_size, void* d_ws, size_t ws_size,
                              hipStream_t stream) {
  (void)in_sizes; (void)n_in; (void)out_size; (void)ws_size;
  const float*         x      = (const float*)d_in[0];
  const int*           tsp    = (const int*)d_in[1];
  /* d_in[2] causal_mask recomputed analytically */
  const unsigned char* pad    = (const unsigned char*)d_in[3];
  const float*         uvqk   = (const float*)d_in[4];
  const float*         o_w    = (const float*)d_in[5];
  const float*         o_b    = (const float*)d_in[6];
  const float*         ln_x_w = (const float*)d_in[7];
  const float*         ln_x_b = (const float*)d_in[8];
  const float*         ln_a_w = (const float*)d_in[9];
  const float*         ln_a_b = (const float*)d_in[10];
  const float*         ts_w   = (const float*)d_in[11];
  const float*         pos_w  = (const float*)d_in[12];

  const size_t M = (size_t)BB * NN;   // 8192 rows
  char* ws = (char*)d_ws;
  size_t off = 0;
  _Float16* xn   = (_Float16*)(ws + off); off += M * DD * sizeof(_Float16);
  _Float16* qf   = (_Float16*)(ws + off); off += M * DD * sizeof(_Float16);
  _Float16* kfb  = (_Float16*)(ws + off); off += M * DD * sizeof(_Float16);
  _Float16* vt   = (_Float16*)(ws + off); off += M * DD * sizeof(_Float16);
  float*    ubuf = (float*)(ws + off);    off += M * DD * sizeof(float);
  float*    attn = (float*)(ws + off);    off += M * DD * sizeof(float);
  _Float16* oin  = (_Float16*)(ws + off); off += M * DD * sizeof(_Float16);
  _Float16* wqt  = (_Float16*)(ws + off); off += (size_t)EE * DD * sizeof(_Float16);
  _Float16* wot  = (_Float16*)(ws + off); off += (size_t)DD * DD * sizeof(_Float16);

  // 0) weight transposes (f32 [K][N] -> f16 [N][K]); L2-resident thereafter
  transpose_f32f16_kernel<<<dim3(EE / 64, DD / 64), 256, 0, stream>>>(uvqk, wqt, DD, EE);
  transpose_f32f16_kernel<<<dim3(DD / 64, DD / 64), 256, 0, stream>>>(o_w, wot, DD, DD);
  // 1) xn = LN(x)
  ln_rows_kernel<<<dim3((unsigned)M), 256, 0, stream>>>(x, ln_x_w, ln_x_b, nullptr, xn);
  // 2) [u|v|q|k] = silu(xn @ uvqk), routed
  uvqk_gemm_kernel<<<dim3((unsigned)(M / 64), EE / 128), 256, 0, stream>>>(
      xn, wqt, ubuf, qf, kfb, vt);
  // 3) fused causal attention with rel-bias
  attn_kernel<<<dim3(NN / 64, HH, BB), 128, 0, stream>>>(
      qf, kfb, vt, tsp, pad, ts_w, pos_w, attn);
  // 4) oin = u * LN(attn)
  ln_rows_kernel<<<dim3((unsigned)M), 256, 0, stream>>>(attn, ln_a_w, ln_a_b, ubuf, oin);
  // 5) out = oin @ o_w + o_b + x, pad-masked
  out_gemm_kernel<<<dim3((unsigned)(M / 64), DD / 128), 256, 0, stream>>>(
      oin, wot, o_b, x, pad, (float*)d_out);
}
